// RegionHead_57939108823106
// MI455X (gfx1250) — matrix operand order repaired and checked by reading, not compiled, not run on hardware
//
#include <hip/hip_runtime.h>
#include <hip/hip_bf16.h>

// ---------------- types ----------------
typedef __attribute__((ext_vector_type(16))) __bf16        v16bf;
typedef __attribute__((ext_vector_type(8)))  float         v8f;
typedef __attribute__((ext_vector_type(4)))  unsigned int  v4u;
typedef __attribute__((ext_vector_type(4)))  unsigned int  u32x4;
typedef __attribute__((ext_vector_type(8)))  int           i32x8;
typedef __attribute__((ext_vector_type(4)))  int           i32x4;

union FragU { v4u u[2]; v16bf h; };   // 32 bytes

// ---------------- problem constants ----------------
#define B_    16
#define T_    8
#define NB_   4
#define CIN   2048
#define CI    512
#define H_    14
#define W_    14
#define THW   (T_ * H_ * W_)        // 1568
#define M1    (B_ * T_ * H_ * W_)   // 25088
#define K2    (CI * 9)              // 4608
#define M2    (B_ * T_ * NB_)       // 512
#define NCLS  174
#define NOBJ  301

#define LDS_PITCH 72                // 64 K + 8 pad halves; 144 B rows, 16B aligned, bank-conflict free

__device__ __forceinline__ unsigned short f2bf(float f) {
    unsigned int u = __float_as_uint(f);
    unsigned int r = u + 0x7FFFu + ((u >> 16) & 1u);   // round-to-nearest-even
    return (unsigned short)(r >> 16);
}

// ---------------- TDM: async 2-D bf16 tile -> LDS (ISA §8) ----------------
// tile_d1 rows of tile_d0 halves; row stride d0_stride halves.
// LDS padding: +4 DWORDs (8 halves) after every 32 DWORDs (64 halves) -> 72-half pitch.
__device__ __forceinline__ void tdm_load_tile_bf16(
    const unsigned short* gsrc, unsigned lds_addr,
    unsigned tensor_d0, unsigned tensor_d1, unsigned long long d0_stride,
    unsigned tile_d0, unsigned tile_d1)
{
    unsigned long long ga = (unsigned long long)(size_t)gsrc;
    u32x4 g0;
    g0[0] = 1u;                                                 // count=1 (valid user D#)
    g0[1] = lds_addr;                                           // LDS byte address
    g0[2] = (unsigned)(ga & 0xFFFFFFFFu);                       // global_addr[31:0]
    g0[3] = (unsigned)((ga >> 32) & 0x01FFFFFFu) | (2u << 30);  // global_addr[56:32] | type=2
    i32x8 g1;
    g1[0] = (int)((1u << 16)      // data_size = 1 -> 2 bytes
                | (1u << 20)      // pad_enable
                | (4u << 22)      // pad_interval: 32 DWORDs (= 64 halves)
                | (3u << 25));    // pad_amount:   4 DWORDs (=  8 halves)
    g1[1] = (int)((tensor_d0 & 0xFFFFu) << 16);                                   // dim0[15:0]
    g1[2] = (int)(((tensor_d0 >> 16) & 0xFFFFu) | ((tensor_d1 & 0xFFFFu) << 16)); // dim0[31:16]|dim1[15:0]
    g1[3] = (int)(((tensor_d1 >> 16) & 0xFFFFu) | ((tile_d0 & 0xFFFFu) << 16));   // dim1[31:16]|tile_dim0
    g1[4] = (int)(tile_d1 & 0xFFFFu);                                             // tile_dim1, tile_dim2=0
    g1[5] = (int)(unsigned)(d0_stride & 0xFFFFFFFFu);                             // dim0_stride[31:0]
    g1[6] = (int)(unsigned)((d0_stride >> 32) & 0xFFFFu);                         // dim0_stride[47:32]
    g1[7] = 0;
    i32x4 z4 = {0, 0, 0, 0};                                    // groups 2/3 unused (2-D tile)
    i32x8 z8 = {0, 0, 0, 0, 0, 0, 0, 0};
    __builtin_amdgcn_tensor_load_to_lds(g0, g1, z4, z4, z8, 0); // 6-arg clang-23 form
}

// ================= converters =================
__global__ __launch_bounds__(256)
void cvt_bf16_kernel(const float* __restrict__ in, unsigned short* __restrict__ out, int n) {
    int i = blockIdx.x * 256 + threadIdx.x;
    if (i < n) out[i] = f2bf(in[i]);
}

// in: [K][N] fp32 (N contiguous) -> out: [N][K] bf16 (K contiguous)
__global__ __launch_bounds__(256)
void transpose_bf16_kernel(const float* __restrict__ in, unsigned short* __restrict__ out,
                           int K, int N) {
    int i = blockIdx.x * 256 + threadIdx.x;
    if (i >= K * N) return;
    int k = i % K, n = i / K;
    out[(size_t)n * K + k] = f2bf(in[(size_t)k * N + n]);
}

// ---------------- shared WMMA micro-kernel ----------------
// Wave tile: 32 rows x 64 cols = 2x4 fragments; WG tile 128x128; K-chunk 64 (2 WMMA K-steps).
__device__ __forceinline__ void frag_mma_stage(
    const unsigned short (*As)[LDS_PITCH], const unsigned short (*Bs)[LDS_PITCH],
    int wm, int wn, int lhalf, int l16, v8f acc[2][4]) {
#pragma unroll
    for (int ks = 0; ks < 64; ks += 32) {
        v16bf af[2], bfr[4];
#pragma unroll
        for (int r = 0; r < 2; ++r) {          // A 16x32: 16 contiguous K halves per lane
            int row = wm * 32 + r * 16 + l16;
            FragU u;
            u.u[0] = *(const v4u*)&As[row][ks + lhalf * 16];
            u.u[1] = *(const v4u*)&As[row][ks + lhalf * 16 + 8];
            af[r] = u.h;
        }
#pragma unroll
        for (int c = 0; c < 4; ++c) {          // B 32x16: 16 contiguous K halves per lane
            int col = wn * 64 + c * 16 + l16;
            FragU u;
            u.u[0] = *(const v4u*)&Bs[col][ks + lhalf * 16];
            u.u[1] = *(const v4u*)&Bs[col][ks + lhalf * 16 + 8];
            bfr[c] = u.h;
        }
#pragma unroll
        for (int r = 0; r < 2; ++r)
#pragma unroll
            for (int c = 0; c < 4; ++c)
                acc[r][c] = __builtin_amdgcn_wmma_f32_16x16x32_bf16(
                    false, af[r], false, bfr[c], (short)0, acc[r][c], false, false);
    }
}

// ================= GEMM1: fm[m][o] = sum_c x[b][c][thw] * w5bf[o][c] =================
// M=25088, N=512, K=2048.  A staged by VALU (fp32->bf16), B tile via TDM.
__global__ __launch_bounds__(256)
void gemm_conv5(const float* __restrict__ x, const unsigned short* __restrict__ wbf,
                float* __restrict__ fm) {
    const int m0 = blockIdx.x * 128;
    const int n0 = blockIdx.y * 128;
    __shared__ __align__(16) unsigned short As[128][LDS_PITCH];
    __shared__ __align__(16) unsigned short Bs[128][LDS_PITCH];

    const int tid   = threadIdx.x;
    const int wave  = tid >> 5;
    const int lane  = tid & 31;
    const int wm    = wave >> 1;
    const int wn    = wave & 1;
    const int lhalf = lane >> 4;
    const int l16   = lane & 15;
    const unsigned bs_lds = (unsigned)(size_t)(void*)&Bs[0][0];

    v8f acc[2][4];
#pragma unroll
    for (int r = 0; r < 2; ++r)
#pragma unroll
        for (int c = 0; c < 4; ++c)
#pragma unroll
            for (int j = 0; j < 8; ++j) acc[r][c][j] = 0.f;

    for (int k0 = 0; k0 < CIN; k0 += 64) {
        // B tile 128(N) x 64(K) halves via TDM (overlaps with A conversion below)
        if (wave == 0)
            tdm_load_tile_bf16(wbf + (size_t)n0 * CIN + k0, bs_lds,
                               CIN - k0, 128u + (unsigned)n0, (unsigned long long)CIN,
                               64u, 128u);
        // A tile 128x64 fp32 -> bf16; 4096 k-pairs, 16 pairs/thread. Coalesced along m.
#pragma unroll
        for (int i = 0; i < 16; ++i) {
            int lin = i * 256 + tid;
            int ml  = lin & 127;
            int kp  = lin >> 7;                  // 0..31 pair index
            int m   = m0 + ml;
            int b   = m / THW;
            int thw = m - b * THW;
            const float* px = x + ((size_t)b * CIN + (k0 + 2 * kp)) * THW + thw;
            unsigned int p = (unsigned int)f2bf(px[0]) |
                             ((unsigned int)f2bf(px[THW]) << 16);
            *(unsigned int*)&As[ml][2 * kp] = p;
        }
        if (wave == 0) __builtin_amdgcn_s_wait_tensorcnt(0);
        __syncthreads();
        frag_mma_stage(As, Bs, wm, wn, lhalf, l16, acc);
        __syncthreads();
    }

    // store C: VGPR j, lane L -> row = j + 8*(L>>4), col = L&15
#pragma unroll
    for (int r = 0; r < 2; ++r)
#pragma unroll
        for (int c = 0; c < 4; ++c)
#pragma unroll
            for (int j = 0; j < 8; ++j) {
                int row = m0 + wm * 32 + r * 16 + j + 8 * lhalf;
                int col = n0 + wn * 64 + c * 16 + l16;
                fm[(size_t)row * CI + col] = acc[r][c][j];
            }
}

// ================= GEMM2: reg = relu(feats_bf16 @ re_wT + re_b) =================
// M=512, N=512, K=4608. A,B both bf16 [row][K]; both tiles via TDM.
__global__ __launch_bounds__(256)
void gemm_region(const unsigned short* __restrict__ A, const unsigned short* __restrict__ Bm,
                 const float* __restrict__ bias, float* __restrict__ out) {
    const int m0 = blockIdx.x * 128;
    const int n0 = blockIdx.y * 128;
    __shared__ __align__(16) unsigned short As[128][LDS_PITCH];
    __shared__ __align__(16) unsigned short Bs[128][LDS_PITCH];

    const int tid   = threadIdx.x;
    const int wave  = tid >> 5;
    const int lane  = tid & 31;
    const int wm    = wave >> 1;
    const int wn    = wave & 1;
    const int lhalf = lane >> 4;
    const int l16   = lane & 15;
    const unsigned as_lds = (unsigned)(size_t)(void*)&As[0][0];
    const unsigned bs_lds = (unsigned)(size_t)(void*)&Bs[0][0];

    v8f acc[2][4];
#pragma unroll
    for (int r = 0; r < 2; ++r)
#pragma unroll
        for (int c = 0; c < 4; ++c)
#pragma unroll
            for (int j = 0; j < 8; ++j) acc[r][c][j] = 0.f;

    for (int k0 = 0; k0 < K2; k0 += 64) {
        if (wave == 0) {
            tdm_load_tile_bf16(A  + (size_t)m0 * K2 + k0, as_lds,
                               K2 - k0, 128u + (unsigned)m0, (unsigned long long)K2,
                               64u, 128u);
            tdm_load_tile_bf16(Bm + (size_t)n0 * K2 + k0, bs_lds,
                               K2 - k0, 128u + (unsigned)n0, (unsigned long long)K2,
                               64u, 128u);
            __builtin_amdgcn_s_wait_tensorcnt(0);
        }
        __syncthreads();
        frag_mma_stage(As, Bs, wm, wn, lhalf, l16, acc);
        __syncthreads();
    }

#pragma unroll
    for (int r = 0; r < 2; ++r)
#pragma unroll
        for (int c = 0; c < 4; ++c)
#pragma unroll
            for (int j = 0; j < 8; ++j) {
                int row = m0 + wm * 32 + r * 16 + j + 8 * lhalf;
                int col = n0 + wn * 64 + c * 16 + l16;
                float v = acc[r][c][j] + bias[col];
                out[(size_t)row * CI + col] = v > 0.f ? v : 0.f;
            }
}

// ================= ROI align =================
__global__ __launch_bounds__(256)
void roi_align_kernel(const float* __restrict__ fm, const float* __restrict__ boxes,
                      unsigned short* __restrict__ feats) {
    const int roi = blockIdx.x;            // bt*NB + nb
    const int bt  = roi >> 2;
    const float* bx = boxes + (size_t)roi * 4;
    const float cx = bx[0], cy = bx[1], w = bx[2], h = bx[3];
    const float x1 = (cx - 0.5f * w) * 14.f;
    const float y1 = (cy - 0.5f * h) * 14.f;
    const float x2 = (cx + 0.5f * w) * 14.f;
    const float y2 = (cy + 0.5f * h) * 14.f;
    const float rw = fmaxf(x2 - x1, 1.f), rh = fmaxf(y2 - y1, 1.f);
    const float bw = rw * (1.f / 3.f), bh = rh * (1.f / 3.f);

    int   y0a[6], y1a[6], x0a[6], x1a[6];
    float lya[6], hya[6], lxa[6], hxa[6];
    bool  vya[6], vxa[6];
#pragma unroll
    for (int i = 0; i < 6; ++i) {
        int p = i >> 1, s = i & 1;
        float yy = y1 + p * bh + (s + 0.5f) * bh * 0.5f;
        float xx = x1 + p * bw + (s + 0.5f) * bw * 0.5f;
        vya[i] = (yy > -1.f) && (yy < 14.f);
        vxa[i] = (xx > -1.f) && (xx < 14.f);
        float y = fminf(fmaxf(yy, 0.f), 13.f);
        float x = fminf(fmaxf(xx, 0.f), 13.f);
        int y0 = (int)floorf(y), x0 = (int)floorf(x);
        y0a[i] = y0; x0a[i] = x0;
        y1a[i] = (y0 + 1 < 13) ? y0 + 1 : 13;
        x1a[i] = (x0 + 1 < 13) ? x0 + 1 : 13;
        lya[i] = y - (float)y0; hya[i] = 1.f - lya[i];
        lxa[i] = x - (float)x0; hxa[i] = 1.f - lxa[i];
    }

    const float* f = fm + (size_t)bt * (H_ * W_) * CI;
    for (int c = threadIdx.x; c < CI; c += 256) {
#pragma unroll
        for (int ph = 0; ph < 3; ++ph)
#pragma unroll
            for (int pw = 0; pw < 3; ++pw) {
                float s = 0.f;
#pragma unroll
                for (int sy = 0; sy < 2; ++sy)
#pragma unroll
                    for (int sx = 0; sx < 2; ++sx) {
                        int iy = ph * 2 + sy, ix = pw * 2 + sx;
                        if (vya[iy] && vxa[ix]) {
                            float v00 = f[(size_t)(y0a[iy] * W_ + x0a[ix]) * CI + c];
                            float v01 = f[(size_t)(y0a[iy] * W_ + x1a[ix]) * CI + c];
                            float v10 = f[(size_t)(y1a[iy] * W_ + x0a[ix]) * CI + c];
                            float v11 = f[(size_t)(y1a[iy] * W_ + x1a[ix]) * CI + c];
                            s += hya[iy] * hxa[ix] * v00 + hya[iy] * lxa[ix] * v01 +
                                 lya[iy] * hxa[ix] * v10 + lya[iy] * lxa[ix] * v11;
                        }
                    }
                feats[(size_t)roi * K2 + c * 9 + ph * 3 + pw] = f2bf(s * 0.25f);
            }
    }
}

// ================= means / gather / heads =================
__global__ __launch_bounds__(256)
void mean_t_kernel(const float* __restrict__ reg, float* __restrict__ reg_mean) {
    int idx = blockIdx.x * 256 + threadIdx.x;   // B*NB*CI = 32768
    if (idx >= B_ * NB_ * CI) return;
    int o = idx % CI, nb = (idx / CI) % NB_, b = idx / (CI * NB_);
    float s = 0.f;
#pragma unroll
    for (int t = 0; t < T_; ++t)
        s += reg[(size_t)((b * T_ + t) * NB_ + nb) * CI + o];
    reg_mean[idx] = s * (1.f / (float)T_);
}

__global__ __launch_bounds__(256)
void mean_nb_kernel(const float* __restrict__ reg_mean, float* __restrict__ pooled) {
    int idx = blockIdx.x * 256 + threadIdx.x;   // B*CI = 8192
    if (idx >= B_ * CI) return;
    int o = idx % CI, b = idx / CI;
    float s = 0.f;
#pragma unroll
    for (int nb = 0; nb < NB_; ++nb) s += reg_mean[(size_t)(b * NB_ + nb) * CI + o];
    pooled[idx] = s * 0.25f;
}

__global__ __launch_bounds__(256)
void obj_gather_kernel(const float* __restrict__ reg_mean, const int* __restrict__ obj_cat,
                       float* __restrict__ obj_feas, float* __restrict__ out_labels) {
    int idx = blockIdx.x * 256 + threadIdx.x;   // 32*512
    if (idx >= B_ * 2 * CI) return;
    int o = idx % CI, r = idx / CI, b = r >> 1, j = r & 1;
    int c0 = obj_cat[b * NB_ + 2];
    int c1 = obj_cat[b * NB_ + 3];
    int sel = (c0 == 0 && c1 != 0) ? (1 - j) : j;   // stable argsort of (cat==0)
    obj_feas[idx] = reg_mean[(size_t)(b * NB_ + 2 + sel) * CI + o];
    if (o == 0) out_labels[r] = (float)(sel == 0 ? c0 : c1);
}

// out[m][n] = sum_k A[m][k] * W[k][n] + bias[n]
__global__ __launch_bounds__(256)
void mm_bias_kernel(const float* __restrict__ A, const float* __restrict__ W,
                    const float* __restrict__ bias, float* __restrict__ out,
                    int M, int N, int K) {
    int idx = blockIdx.x * 256 + threadIdx.x;
    if (idx >= M * N) return;
    int n = idx % N, m = idx / N;
    float s = bias[n];
    for (int k = 0; k < K; ++k) s += A[(size_t)m * K + k] * W[(size_t)k * N + n];
    out[idx] = s;
}

// ================= host orchestration =================
extern "C" void kernel_launch(void* const* d_in, const int* in_sizes, int n_in,
                              void* d_out, int out_size, void* d_ws, size_t ws_size,
                              hipStream_t stream) {
    const float* x      = (const float*)d_in[0];
    const float* boxes  = (const float*)d_in[1];
    const int*   cat    = (const int*)d_in[2];
    const float* w5     = (const float*)d_in[3];
    const float* re_w   = (const float*)d_in[4];
    const float* re_b   = (const float*)d_in[5];
    const float* oc1_w  = (const float*)d_in[6];
    const float* oc1_b  = (const float*)d_in[7];
    const float* oc2_w  = (const float*)d_in[8];
    const float* oc2_b  = (const float*)d_in[9];
    const float* pr1_w  = (const float*)d_in[10];
    const float* pr1_b  = (const float*)d_in[11];
    const float* pr2_w  = (const float*)d_in[12];
    const float* pr2_b  = (const float*)d_in[13];
    float* out = (float*)d_out;

    char* ws = (char*)d_ws;
    size_t off = 0;
    unsigned short* w5_bf   = (unsigned short*)(ws + off); off += (size_t)CI * CIN * 2;
    unsigned short* rewT_bf = (unsigned short*)(ws + off); off += (size_t)CI * K2 * 2;
    float*          fm      = (float*)(ws + off);          off += (size_t)M1 * CI * 4;
    unsigned short* feats   = (unsigned short*)(ws + off); off += (size_t)M2 * K2 * 2;
    float*          reg     = (float*)(ws + off);          off += (size_t)M2 * CI * 4;
    float*          regmean = (float*)(ws + off);          off += (size_t)B_ * NB_ * CI * 4;
    float*          pooled  = (float*)(ws + off);          off += (size_t)B_ * CI * 4;
    float*          objfeas = (float*)(ws + off);          off += (size_t)B_ * 2 * CI * 4;
    float*          h_obj   = (float*)(ws + off);          off += (size_t)B_ * 2 * CI * 4;
    float*          h_pr    = (float*)(ws + off);          off += (size_t)B_ * CI * 4;

    // 0) one-pass weight conversions
    cvt_bf16_kernel<<<(CI * CIN + 255) / 256, 256, 0, stream>>>(w5, w5_bf, CI * CIN);
    transpose_bf16_kernel<<<(K2 * CI + 255) / 256, 256, 0, stream>>>(re_w, rewT_bf, K2, CI);

    // 1) conv5 GEMM (WMMA bf16, TDM-staged B): fm[bthw][512]
    gemm_conv5<<<dim3(M1 / 128, CI / 128), 256, 0, stream>>>(x, w5_bf, fm);

    // 2) ROI align -> feats bf16 [512][4608]
    roi_align_kernel<<<M2, 256, 0, stream>>>(fm, boxes, feats);

    // 3) region GEMM (WMMA bf16, TDM-staged A+B) + bias + relu: reg [512][512]
    gemm_region<<<dim3(M2 / 128, CI / 128), 256, 0, stream>>>(feats, rewT_bf, re_b, reg);

    // 4) reductions
    mean_t_kernel<<<(B_ * NB_ * CI + 255) / 256, 256, 0, stream>>>(reg, regmean);
    mean_nb_kernel<<<(B_ * CI + 255) / 256, 256, 0, stream>>>(regmean, pooled);

    // 5) object branch (labels written straight into d_out tail)
    float* out_cls    = out;                         // 16*174
    float* out_objcls = out + B_ * NCLS;             // 32*301
    float* out_labels = out_objcls + B_ * 2 * NOBJ;  // 32
    obj_gather_kernel<<<(B_ * 2 * CI + 255) / 256, 256, 0, stream>>>(regmean, cat, objfeas, out_labels);
    mm_bias_kernel<<<(B_ * 2 * CI + 255) / 256, 256, 0, stream>>>(objfeas, oc1_w, oc1_b, h_obj, B_ * 2, CI, CI);
    mm_bias_kernel<<<(B_ * 2 * NOBJ + 255) / 256, 256, 0, stream>>>(h_obj, oc2_w, oc2_b, out_objcls, B_ * 2, NOBJ, CI);

    // 6) classification branch
    mm_bias_kernel<<<(B_ * CI + 255) / 256, 256, 0, stream>>>(pooled, pr1_w, pr1_b, h_pr, B_, CI, CI);
    mm_bias_kernel<<<(B_ * NCLS + 255) / 256, 256, 0, stream>>>(h_pr, pr2_w, pr2_b, out_cls, B_, NCLS, CI);

    (void)in_sizes; (void)n_in; (void)out_size; (void)ws_size;
}
